// BiLSTMCRF_80195629351539
// MI455X (gfx1250) — matrix operand order repaired and checked by reading, not compiled, not run on hardware
//
#include <hip/hip_runtime.h>

// ---------------------------------------------------------------------------
// BiLSTM-CRF for MI455X (gfx1250, wave32).
//  - Time-batched projections: bf16 WMMA GEMMs (v_wmma_f32_16x16x32_bf16).
//  - LSTM recurrence: 8 workgroups per direction, each holding its 256KB
//    recurrent-weight slice in LDS (320KB/WGP); per-step cross-WG sync via
//    release atomicAdd + acquire spin on a monotonic counter, double-buffered
//    global h vector (parity by step => no >1-step skew possible).
//  - GRU recurrence: single WG, 192KB Whh LDS-resident.
//  - Viterbi DP: 16-lane forward pass + single-thread backtrack.
// ---------------------------------------------------------------------------

typedef __attribute__((ext_vector_type(16))) __bf16 v16bf;
typedef __attribute__((ext_vector_type(8)))  float  v8f;

#define T_LEN 4096
#define ED    256
#define SED   128
#define SHD   128
#define HD    512
#define KTAGS 16
#define START_TAG 14
#define STOP_TAG  15
#define NEGV  (-10000.0f)

#define LSTM_G   8          // workgroups per direction
#define LSTM_CH  64         // h elements per workgroup
#define WROW_DW  257        // padded LDS row stride (dwords): bank = row%64
#define GRU_PAD  129        // padded GRU LDS row stride (dwords)

// ---- helpers ---------------------------------------------------------------
__device__ __forceinline__ unsigned short f2bf(float f) {
  unsigned int u = __float_as_uint(f);
  unsigned int r = u + 0x7FFFu + ((u >> 16) & 1u);   // round-to-nearest-even
  return (unsigned short)(r >> 16);
}
__device__ __forceinline__ float bfl(unsigned int bits16) {
  return __uint_as_float(bits16 << 16);
}
__device__ __forceinline__ float sigm(float x) { return 1.0f / (1.0f + expf(-x)); }

// ---- f32 -> bf16 weight conversion ----------------------------------------
__global__ __launch_bounds__(256) void cvt_f32_bf16(const float* __restrict__ s,
                                                    unsigned short* __restrict__ d, int n) {
  int i = blockIdx.x * blockDim.x + threadIdx.x;
  if (i < n) d[i] = f2bf(s[i]);
}

// ---- embedding gather (E -> x0 cols [0,256), SubE -> sube) ----------------
__global__ __launch_bounds__(256) void gather_embed(const int* __restrict__ x,
                                                    const float* __restrict__ E,
                                                    const float* __restrict__ SubE,
                                                    unsigned short* __restrict__ x0, int ld0,
                                                    unsigned short* __restrict__ sube) {
  int t = blockIdx.x;
  int j = threadIdx.x;          // 256 threads
  int id = x[t];
  x0[(size_t)t * ld0 + j] = f2bf(E[(size_t)id * ED + j]);
  if (j < SED) sube[(size_t)t * SED + j] = f2bf(SubE[(size_t)id * SED + j]);
}

// ---- bf16 WMMA GEMM:  out[M,N] = A[M,K] * W[N,K]^T + bias[N]  (f32 out) ----
__global__ __launch_bounds__(256) void wmma_gemm_bias(
    const unsigned short* __restrict__ A, int lda,
    const unsigned short* __restrict__ W, int ldw,
    const float* __restrict__ bias,
    float* __restrict__ out, int ldc,
    int tilesN, int totalTiles, int Kd) {
  int wave = blockIdx.x * (blockDim.x >> 5) + (threadIdx.x >> 5);
  if (wave >= totalTiles) return;                 // uniform per-wave exit (EXEC all-1)
  int tm = wave / tilesN;
  int tn = wave - tm * tilesN;
  int lane = threadIdx.x & 31;
  int half = lane >> 4;
  int r    = lane & 15;

  // A fragment: lane<16 -> K {0..7,16..23}; lane>=16 -> K {8..15,24..31}
  const unsigned short* Arow = A + (size_t)(tm * 16 + r) * lda + half * 8;
  // B fragment: lane holds column n = r; lane<16 -> K {0..7,16..23}; lane>=16 -> K {8..15,24..31}
  const unsigned short* Wrow = W + (size_t)(tn * 16 + r) * ldw + half * 8;

  v8f c = {};
  for (int k = 0; k < Kd; k += 32) {
    union { uint4 u[2]; v16bf v; } a, b;
    a.u[0] = *(const uint4*)(Arow + k);
    a.u[1] = *(const uint4*)(Arow + k + 16);
    b.u[0] = *(const uint4*)(Wrow + k);
    b.u[1] = *(const uint4*)(Wrow + k + 16);
    __builtin_prefetch((const void*)(Arow + k + 32), 0, 0);  // global_prefetch_b8
    c = __builtin_amdgcn_wmma_f32_16x16x32_bf16(false, a.v, false, b.v,
                                                (short)0, c, false, false);
  }

  int n = tn * 16 + r;
  float bv = bias ? bias[n] : 0.0f;
#pragma unroll
  for (int j = 0; j < 8; ++j) {
    int m = tm * 16 + j + half * 8;               // C/D layout: VGPR j -> M = j + 8*half
    out[(size_t)m * ldc + n] = c[j] + bv;
  }
}

// ---- per-phase recurrence init: h0 -> hbuf[0], counter -> 0 ---------------
__global__ __launch_bounds__(512) void init_rec(const float* __restrict__ h0,
                                                float* __restrict__ hbuf,
                                                unsigned int* __restrict__ cnt) {
  int j = threadIdx.x;
  hbuf[j] = h0[j];
  if (j == 0) *cnt = 0u;
}

// ---- GRU recurrence (single WG, 384 threads; Whh f32 LDS-resident) --------
__global__ __launch_bounds__(384) void gru_rec(const float* __restrict__ gi,
                                               const float* __restrict__ Whh,
                                               const float* __restrict__ bhh,
                                               unsigned short* __restrict__ x0, int ld0,
                                               int T) {
  extern __shared__ float sm[];
  float* wlds = sm;                         // 384 * GRU_PAD
  float* h    = wlds + 384 * GRU_PAD;       // 128
  float* gh   = h + SHD;                    // 384
  int j = threadIdx.x;
  // stage Whh into LDS (row j -> padded row j, bank = j%64: conflict-free)
  const float* wsrc = Whh + (size_t)j * SHD;
  float* wrow = wlds + j * GRU_PAD;
  for (int k = 0; k < SHD; ++k) wrow[k] = wsrc[k];
  float bh = bhh[j];
  if (j < SHD) h[j] = 0.0f;                 // zero initial hidden
  __syncthreads();
  for (int t = 0; t < T; ++t) {
    float acc = bh;
#pragma unroll 8
    for (int k = 0; k < SHD; ++k) acc += wrow[k] * h[k];
    gh[j] = acc;                             // gh = h@Whh^T + bhh (gi NOT folded: n-gate)
    __syncthreads();
    if (j < SHD) {
      const float* g = gi + (size_t)t * (3 * SHD);
      float rr = sigm(g[j] + gh[j]);
      float zz = sigm(g[SHD + j] + gh[SHD + j]);
      float nn = tanhf(g[2 * SHD + j] + rr * gh[2 * SHD + j]);
      float hn = (1.0f - zz) * nn + zz * h[j];
      h[j] = hn;
      x0[(size_t)t * ld0 + ED + j] = f2bf(hn);   // concat [emb | gru_out]
    }
    __syncthreads();
  }
}

// ---- LSTM recurrence: 8 WGs/direction, 256KB weight slice in LDS ----------
// gates[t] holds x_t @ Wih^T + bih (f32). WG g owns h[64g..64g+63] and gate
// rows {64g+e, 512+64g+e, 1024+64g+e, 1536+64g+e}. h exchanged through a
// double-buffered global vector; step barrier = monotonic arrival counter.
__global__ __launch_bounds__(256) void lstm_rec_mwg(
    const float* __restrict__ gates,
    const unsigned short* __restrict__ Whh,
    const float* __restrict__ bhh,
    const float* __restrict__ c0,
    float* __restrict__ hbuf,            // 2 * 512 floats (parity double buffer)
    unsigned int* __restrict__ cnt,      // arrival counter (zeroed by init_rec)
    unsigned short* __restrict__ out, int ldo, int colOff, int T, int dir) {
  extern __shared__ unsigned int smem[];
  unsigned int* wlds = smem;                       // 256 * WROW_DW dwords
  float* harr = (float*)(wlds + 256 * WROW_DW);    // 512
  float* gh   = harr + HD;                         // 256

  int g = blockIdx.x;                 // 0..7
  int j = threadIdx.x;                // 0..255
  int gate = j >> 6;                  // 0..3 (i,f,g,o)
  int e    = j & 63;
  int hBase = g * LSTM_CH;
  int grow  = gate * HD + hBase + e;  // global gate row

  // stage this WG's 256 weight rows into LDS, packed 2 bf16/dword,
  // padded row stride WROW_DW => lane row j starts at bank j%64 (conflict-free)
  const unsigned int* wsrc = (const unsigned int*)(Whh + (size_t)grow * HD);
  unsigned int* wrow = wlds + j * WROW_DW;
  for (int kd = 0; kd < HD / 2; ++kd) wrow[kd] = wsrc[kd];
  float bh = bhh[grow];
  float c = (j < LSTM_CH) ? c0[hBase + j] : 0.0f;
  __syncthreads();

  for (int s = 0; s < T; ++s) {
    int t = dir ? (T - 1 - s) : s;
    // pull current h (written to parity buffer s&1) into LDS
    const float* hsrc = hbuf + (s & 1) * HD;
    harr[j]       = hsrc[j];
    harr[256 + j] = hsrc[256 + j];
    __syncthreads();

    float acc = bh + gates[(size_t)t * (4 * HD) + grow];   // fold gi (all gates additive)
#pragma unroll 8
    for (int kd = 0; kd < HD / 2; ++kd) {
      unsigned int u = wrow[kd];
      acc += bfl(u & 0xFFFFu) * harr[2 * kd] + bfl(u >> 16) * harr[2 * kd + 1];
    }
    gh[j] = acc;
    // prefetch next step's gate row while others finish
    int t2 = dir ? (t - 1) : (t + 1);
    if ((unsigned)t2 < (unsigned)T)
      __builtin_prefetch((const void*)&gates[(size_t)t2 * (4 * HD) + grow], 0, 0);
    __syncthreads();

    if (j < LSTM_CH) {
      float gi_ = gh[j];                 // i
      float gf  = gh[LSTM_CH + j];       // f
      float gg  = gh[2 * LSTM_CH + j];   // g
      float go  = gh[3 * LSTM_CH + j];   // o
      c = sigm(gf) * c + sigm(gi_) * tanhf(gg);
      float hn = sigm(go) * tanhf(c);
      out[(size_t)t * ldo + colOff + hBase + j] = f2bf(hn);
      hbuf[((s + 1) & 1) * HD + hBase + j] = hn;   // publish next-step h chunk
    }
    __syncthreads();
    __threadfence();                    // make h chunk visible device-wide
    if (j == 0) {
      __hip_atomic_fetch_add(cnt, 1u, __ATOMIC_RELEASE, __HIP_MEMORY_SCOPE_AGENT);
      unsigned int target = (unsigned int)(LSTM_G) * (unsigned int)(s + 1);
      while (__hip_atomic_load(cnt, __ATOMIC_ACQUIRE, __HIP_MEMORY_SCOPE_AGENT) < target)
        __builtin_amdgcn_s_sleep(2);
    }
    __syncthreads();
  }
}

// ---- Viterbi: 16-lane forward DP + single-thread backtrack -----------------
__global__ __launch_bounds__(256) void viterbi_kernel(const float* __restrict__ feats,
                                                      const float* __restrict__ trans,
                                                      int* __restrict__ bp,
                                                      int* __restrict__ pathOut,
                                                      float* __restrict__ scoreOut, int T) {
  __shared__ float tr[KTAGS * KTAGS];
  __shared__ float fv[KTAGS];
  __shared__ float nfv[KTAGS];
  int j = threadIdx.x;
  if (j < KTAGS * KTAGS) tr[j] = trans[j];
  if (j < KTAGS) fv[j] = (j == START_TAG) ? 0.0f : NEGV;
  __syncthreads();
  for (int t = 0; t < T; ++t) {
    if (j < KTAGS) {
      float best = -3.4e38f; int arg = 0;
#pragma unroll
      for (int p = 0; p < KTAGS; ++p) {
        float s = fv[p] + tr[j * KTAGS + p];
        if (s > best) { best = s; arg = p; }   // strict > == argmax-first-tie
      }
      bp[t * KTAGS + j] = arg;
      nfv[j] = best + feats[(size_t)t * KTAGS + j];
    }
    __syncthreads();
    if (j < KTAGS) fv[j] = nfv[j];
    __syncthreads();
  }
  if (j == 0) {
    float best = -3.4e38f; int bt = 0;
#pragma unroll
    for (int k = 0; k < KTAGS; ++k) {
      float s = fv[k] + tr[STOP_TAG * KTAGS + k];
      if (s > best) { best = s; bt = k; }
    }
    *scoreOut = best;
    int tag = bt;
    for (int t = T - 1; t >= 0; --t) { pathOut[t] = tag; tag = bp[t * KTAGS + tag]; }
  }
}

// ---------------------------------------------------------------------------
static inline void launch_gemm(const unsigned short* A, int lda,
                               const unsigned short* W, int ldw,
                               const float* bias, float* out, int ldc,
                               int M, int N, int K, hipStream_t s) {
  int tiles = (M / 16) * (N / 16);
  int grid = (tiles + 7) / 8;                        // 8 waves (256 thr) per block
  wmma_gemm_bias<<<grid, 256, 0, s>>>(A, lda, W, ldw, bias, out, ldc, N / 16, tiles, K);
}

static inline void launch_cvt(const float* s, unsigned short* d, int n, hipStream_t st) {
  cvt_f32_bf16<<<(n + 255) / 256, 256, 0, st>>>(s, d, n);
}

extern "C" void kernel_launch(void* const* d_in, const int* in_sizes, int n_in,
                              void* d_out, int out_size, void* d_ws, size_t ws_size,
                              hipStream_t stream) {
  (void)in_sizes; (void)n_in; (void)out_size; (void)ws_size;
  const int T = T_LEN;

  // Inputs (setup_inputs dict order, params flattened in insertion order)
  const int*   x        = (const int*)  d_in[0];
  const float* h0       = (const float*)d_in[1];   // (2L,1,HD)
  const float* c0       = (const float*)d_in[2];
  const float* E        = (const float*)d_in[3];
  const float* SubE     = (const float*)d_in[4];
  const float* gru_Wih  = (const float*)d_in[5];   // (384,128)
  const float* gru_Whh  = (const float*)d_in[6];   // (384,128)
  const float* gru_bih  = (const float*)d_in[7];
  const float* gru_bhh  = (const float*)d_in[8];
  const float* l0Wih_f = (const float*)d_in[9];    const float* l0bih_f = (const float*)d_in[11];
  const float* l0Whh_f = (const float*)d_in[10];   const float* l0bhh_f = (const float*)d_in[12];
  const float* l0Wih_b = (const float*)d_in[13];   const float* l0bih_b = (const float*)d_in[15];
  const float* l0Whh_b = (const float*)d_in[14];   const float* l0bhh_b = (const float*)d_in[16];
  const float* l1Wih_f = (const float*)d_in[17];   const float* l1bih_f = (const float*)d_in[19];
  const float* l1Whh_f = (const float*)d_in[18];   const float* l1bhh_f = (const float*)d_in[20];
  const float* l1Wih_b = (const float*)d_in[21];   const float* l1bih_b = (const float*)d_in[23];
  const float* l1Whh_b = (const float*)d_in[22];   const float* l1bhh_b = (const float*)d_in[24];
  const float* W_out   = (const float*)d_in[25];   // (16, 1024)
  const float* b_out   = (const float*)d_in[26];
  const float* trans   = (const float*)d_in[27];   // (16,16)

  // Workspace carve-up
  char* ws = (char*)d_ws;
  size_t off = 0;
  auto alloc = [&](size_t bytes) -> void* {
    void* p = ws + off; off = (off + bytes + 255) & ~(size_t)255; return p;
  };
  unsigned short* gruWih_bf = (unsigned short*)alloc((size_t)384 * 128 * 2);
  unsigned short* l0Wf_bf   = (unsigned short*)alloc((size_t)2048 * 384 * 2);
  unsigned short* l0Wb_bf   = (unsigned short*)alloc((size_t)2048 * 384 * 2);
  unsigned short* l1Wf_bf   = (unsigned short*)alloc((size_t)2048 * 1024 * 2);
  unsigned short* l1Wb_bf   = (unsigned short*)alloc((size_t)2048 * 1024 * 2);
  unsigned short* Whh0f_bf  = (unsigned short*)alloc((size_t)2048 * 512 * 2);
  unsigned short* Whh0b_bf  = (unsigned short*)alloc((size_t)2048 * 512 * 2);
  unsigned short* Whh1f_bf  = (unsigned short*)alloc((size_t)2048 * 512 * 2);
  unsigned short* Whh1b_bf  = (unsigned short*)alloc((size_t)2048 * 512 * 2);
  unsigned short* Wout_bf   = (unsigned short*)alloc((size_t)16 * 1024 * 2);
  unsigned short* sube_bf   = (unsigned short*)alloc((size_t)T * 128 * 2);
  unsigned short* x0_bf     = (unsigned short*)alloc((size_t)T * 384 * 2);   // [emb|gru]
  unsigned short* h1_bf     = (unsigned short*)alloc((size_t)T * 1024 * 2);  // layer-1 out
  unsigned short* h2_bf     = (unsigned short*)alloc((size_t)T * 1024 * 2);  // layer-2 out
  float* gi_gru = (float*)alloc((size_t)T * 384 * 4);
  float* gatesF = (float*)alloc((size_t)T * 2048 * 4);
  float* gatesB = (float*)alloc((size_t)T * 2048 * 4);
  float* feats  = (float*)alloc((size_t)T * 16 * 4);
  int*   bp     = (int*)  alloc((size_t)T * 16 * 4);
  float* hbufF  = (float*)alloc((size_t)2 * HD * 4);   // double-buffered h (fwd)
  float* hbufB  = (float*)alloc((size_t)2 * HD * 4);   // double-buffered h (bwd)
  unsigned int* cntF = (unsigned int*)alloc(256);
  unsigned int* cntB = (unsigned int*)alloc(256);

  // 0) weights -> bf16 (deterministic, every call)
  launch_cvt(gru_Wih, gruWih_bf, 384 * 128, stream);
  launch_cvt(l0Wih_f, l0Wf_bf, 2048 * 384, stream);
  launch_cvt(l0Wih_b, l0Wb_bf, 2048 * 384, stream);
  launch_cvt(l1Wih_f, l1Wf_bf, 2048 * 1024, stream);
  launch_cvt(l1Wih_b, l1Wb_bf, 2048 * 1024, stream);
  launch_cvt(l0Whh_f, Whh0f_bf, 2048 * 512, stream);
  launch_cvt(l0Whh_b, Whh0b_bf, 2048 * 512, stream);
  launch_cvt(l1Whh_f, Whh1f_bf, 2048 * 512, stream);
  launch_cvt(l1Whh_b, Whh1b_bf, 2048 * 512, stream);
  launch_cvt(W_out,   Wout_bf, 16 * 1024, stream);

  // 1) embedding gather
  gather_embed<<<T, 256, 0, stream>>>(x, E, SubE, x0_bf, 384, sube_bf);

  // 2) GRU input projection (WMMA): gi = sube @ gru_Wih^T + bih   (4096x384)
  launch_gemm(sube_bf, 128, gruWih_bf, 128, gru_bih, gi_gru, 384, T, 384, 128, stream);

  // 3) GRU recurrence (Whh LDS-resident) -> x0_bf cols [256,384)
  size_t gruLds = (size_t)(384 * GRU_PAD + SHD + 384) * 4;
  gru_rec<<<1, 384, gruLds, stream>>>(gi_gru, gru_Whh, gru_bhh, x0_bf, 384, T);

  // 4) LSTM layer 0: WMMA input projections, then 8-WG LDS-resident recurrences
  size_t lstmLds = (size_t)(256 * WROW_DW) * 4 + (size_t)(HD + 256) * 4;
  launch_gemm(x0_bf, 384, l0Wf_bf, 384, l0bih_f, gatesF, 2048, T, 2048, 384, stream);
  launch_gemm(x0_bf, 384, l0Wb_bf, 384, l0bih_b, gatesB, 2048, T, 2048, 384, stream);
  init_rec<<<1, 512, 0, stream>>>(h0 + 0 * HD, hbufF, cntF);
  init_rec<<<1, 512, 0, stream>>>(h0 + 1 * HD, hbufB, cntB);
  lstm_rec_mwg<<<LSTM_G, 256, lstmLds, stream>>>(gatesF, Whh0f_bf, l0bhh_f,
                                                 c0 + 0 * HD, hbufF, cntF,
                                                 h1_bf, 1024, 0, T, 0);
  lstm_rec_mwg<<<LSTM_G, 256, lstmLds, stream>>>(gatesB, Whh0b_bf, l0bhh_b,
                                                 c0 + 1 * HD, hbufB, cntB,
                                                 h1_bf, 1024, 512, T, 1);

  // 5) LSTM layer 1
  launch_gemm(h1_bf, 1024, l1Wf_bf, 1024, l1bih_f, gatesF, 2048, T, 2048, 1024, stream);
  launch_gemm(h1_bf, 1024, l1Wb_bf, 1024, l1bih_b, gatesB, 2048, T, 2048, 1024, stream);
  init_rec<<<1, 512, 0, stream>>>(h0 + 2 * HD, hbufF, cntF);
  init_rec<<<1, 512, 0, stream>>>(h0 + 3 * HD, hbufB, cntB);
  lstm_rec_mwg<<<LSTM_G, 256, lstmLds, stream>>>(gatesF, Whh1f_bf, l1bhh_f,
                                                 c0 + 2 * HD, hbufF, cntF,
                                                 h2_bf, 1024, 0, T, 0);
  lstm_rec_mwg<<<LSTM_G, 256, lstmLds, stream>>>(gatesB, Whh1b_bf, l1bhh_b,
                                                 c0 + 3 * HD, hbufB, cntB,
                                                 h2_bf, 1024, 512, T, 1);

  // 6) output projection (WMMA): feats = h2 @ W_out^T + b_out   (4096x16)
  launch_gemm(h2_bf, 1024, Wout_bf, 1024, b_out, feats, 16, T, 16, 1024, stream);

  // 7) Viterbi: path (int32) into d_out[0..T), score (f32) at d_out[T]
  viterbi_kernel<<<1, 256, 0, stream>>>(feats, trans, bp,
                                        (int*)d_out, ((float*)d_out) + T, T);
}
